// _SwinBlock_25718264168507
// MI455X (gfx1250) — matrix operand order repaired and checked by reading, not compiled, not run on hardware
//
#include <hip/hip_runtime.h>
#include <math.h>

typedef _Float16 f16;
typedef __attribute__((ext_vector_type(16))) _Float16 v16h;
typedef __attribute__((ext_vector_type(8)))  _Float16 v8h;
typedef __attribute__((ext_vector_type(8)))  float    v8f;
typedef __attribute__((ext_vector_type(4)))  int      v4i;

union Frag16 { v16h v; v8h h[2]; };

// ---------------------------------------------------------------------------
// Static problem config (Swin block)
// B=16, H=W=64, C=512, HEADS=16, WS=8, SHIFT=4, N=64 tok/window, NW=64,
// HID=2048, HEAD_DIM=32.  Rows (tokens) = 65536.
// ---------------------------------------------------------------------------
#define ROWS      65536
#define CCH       512
#define HIDCH     2048
#define QKVCH     1536
#define NHEADS    16
#define ATT_SCALE 0.1767766952966369f

// ---------------------------------------------------------------------------
// CDNA5 async global->LDS staging (ASYNCcnt-tracked), with fallback.
// Builtin signature (probe-discovered): (int4 AS1* src, int4 AS3* dst, Ii, Ii)
// ---------------------------------------------------------------------------
#if defined(__gfx1250__) && __has_builtin(__builtin_amdgcn_global_load_async_to_lds_b128)
#define HAVE_ASYNC_LDS 1
#endif

__device__ __forceinline__ void async_wait0() {
#if __has_builtin(__builtin_amdgcn_s_wait_asynccnt)
  __builtin_amdgcn_s_wait_asynccnt(0);
#else
  asm volatile("s_wait_asynccnt 0" ::: "memory");
#endif
}

#ifdef HAVE_ASYNC_LDS
typedef __attribute__((address_space(1))) v4i ga_v4i;   // global
typedef __attribute__((address_space(3))) v4i ls_v4i;   // LDS

__device__ __forceinline__ void async_copy16(const f16* g, f16* l) {
  // generic->AS(1)/AS(3) via integer cast; flat-LDS low 32 bits == LDS offset.
  __builtin_amdgcn_global_load_async_to_lds_b128(
      (ga_v4i*)(unsigned long long)g,
      (ls_v4i*)(unsigned)(unsigned long long)l,
      0, 0);
}
#endif

// ---------------------------------------------------------------------------
// fp32 -> f16 conversion (weights)
// ---------------------------------------------------------------------------
__global__ __launch_bounds__(256) void cvt_f32_f16(const float* __restrict__ s,
                                                   f16* __restrict__ d, int n) {
  int i = blockIdx.x * 256 + threadIdx.x;
  if (i < n) d[i] = (f16)s[i];
}

// ---------------------------------------------------------------------------
// Pre-gather relative position bias: biasHN[head][n*64+m]
// ---------------------------------------------------------------------------
__global__ __launch_bounds__(256) void bias_pre(const float* __restrict__ table,
                                                const int* __restrict__ idx,
                                                float* __restrict__ biasHN) {
  int e = blockIdx.x * 256 + threadIdx.x;      // 65536 = 16 heads * 4096
  int head = e >> 12;
  int nm = e & 4095;
  biasHN[head * 4096 + nm] = table[idx[nm] * NHEADS + head];
}

// ---------------------------------------------------------------------------
// LayerNorm over C=512, one wave per row.  gather=1 applies the shifted-window
// partition mapping (output row is window-space, input row is image-space).
// ---------------------------------------------------------------------------
__global__ __launch_bounds__(256) void ln_kernel(const float* __restrict__ X,
                                                 const float* __restrict__ g,
                                                 const float* __restrict__ b,
                                                 f16* __restrict__ Y, int gather) {
  int lane = threadIdx.x & 31;
  int wave = threadIdx.x >> 5;
  long orow = (long)blockIdx.x * 8 + wave;
  long irow = orow;
  if (gather) {
    long bb = orow >> 12;
    int rem = (int)(orow & 4095);
    int wi = rem >> 6, t = rem & 63;
    int hs = (wi >> 3) * 8 + (t >> 3);
    int wsx = (wi & 7) * 8 + (t & 7);
    int h = (hs + 4) & 63, w = (wsx + 4) & 63;
    irow = (bb << 12) + h * 64 + w;
  }
  const float4* x4 = (const float4*)(X + irow * CCH);
  float4 v[4];
  float s = 0.f, s2 = 0.f;
#pragma unroll
  for (int i = 0; i < 4; ++i) {
    v[i] = x4[lane + i * 32];
    s += v[i].x + v[i].y + v[i].z + v[i].w;
    s2 += v[i].x * v[i].x + v[i].y * v[i].y + v[i].z * v[i].z + v[i].w * v[i].w;
  }
#pragma unroll
  for (int m = 1; m < 32; m <<= 1) {
    s  += __shfl_xor(s, m, 32);
    s2 += __shfl_xor(s2, m, 32);
  }
  float mean = s * (1.f / 512.f);
  float rstd = rsqrtf(s2 * (1.f / 512.f) - mean * mean + 1e-5f);
  const float4* g4 = (const float4*)g;
  const float4* b4 = (const float4*)b;
  f16* y = Y + orow * CCH;
#pragma unroll
  for (int i = 0; i < 4; ++i) {
    int c4 = lane + i * 32;
    float4 gg = g4[c4], bb2 = b4[c4];
    y[c4 * 4 + 0] = (f16)((v[i].x - mean) * rstd * gg.x + bb2.x);
    y[c4 * 4 + 1] = (f16)((v[i].y - mean) * rstd * gg.y + bb2.y);
    y[c4 * 4 + 2] = (f16)((v[i].z - mean) * rstd * gg.z + bb2.z);
    y[c4 * 4 + 3] = (f16)((v[i].w - mean) * rstd * gg.w + bb2.w);
  }
}

// ---------------------------------------------------------------------------
// Generic WMMA GEMM: out = epilogue( A(MxK,f16) * W(NxK,f16)^T + bias )
// MODE 0: store f16             MODE 1: GELU(exact) -> f16
// MODE 2: out1 = x + acc, scattered via shifted-window reverse map (f32)
// MODE 3: out = resid + acc (f32, identity rows)
// Block: 256 thr (8 waves), tile 128x128, K-step 32.
// Tiles staged to LDS with GLOBAL_LOAD_ASYNC_TO_LDS_B128 when available.
// ---------------------------------------------------------------------------
#define LDSTR 56   // padded LDS row stride in halves (112B = 7*16B)

template <int MODE>
__global__ __launch_bounds__(256) void gemm_wmma(const f16* __restrict__ A,
                                                 const f16* __restrict__ W,
                                                 const float* __restrict__ bias,
                                                 const float* __restrict__ resid,
                                                 void* __restrict__ out,
                                                 int M, int N, int K) {
  __shared__ f16 As[128][LDSTR];
  __shared__ f16 Bs[128][LDSTR];
  const int tid = threadIdx.x;
  const int lane = tid & 31;
  const int wave = tid >> 5;
  const int wm = wave & 1;   // 64-row sub-block
  const int wn = wave >> 1;  // 32-col sub-block
  const long m0 = (long)blockIdx.y * 128;
  const long n0 = (long)blockIdx.x * 128;

  v8f acc[4][2];
#pragma unroll
  for (int i = 0; i < 4; ++i)
#pragma unroll
    for (int j = 0; j < 2; ++j)
#pragma unroll
      for (int p = 0; p < 8; ++p) acc[i][j][p] = 0.f;

  const int ldRow = tid >> 2;         // 0..63
  const int ldSeg = (tid & 3) * 8;    // halves: 0,8,16,24
  const int aoff = (lane < 16) ? 0 : 8;
  const int boff = (lane < 16) ? 0 : 8;

  for (int k0 = 0; k0 < K; k0 += 32) {
#pragma unroll
    for (int it = 0; it < 2; ++it) {
      int r = ldRow + it * 64;
      const f16* gA = A + (m0 + r) * (long)K + k0 + ldSeg;
      const f16* gB = W + (n0 + r) * (long)K + k0 + ldSeg;
#ifdef HAVE_ASYNC_LDS
      async_copy16(gA, &As[r][ldSeg]);
      async_copy16(gB, &Bs[r][ldSeg]);
#else
      *(v8h*)&As[r][ldSeg] = *(const v8h*)gA;
      *(v8h*)&Bs[r][ldSeg] = *(const v8h*)gB;
      if (k0 + 32 < K) {   // prefetch next tiles (global_prefetch_b8)
        __builtin_prefetch(gA + 32, 0, 1);
        __builtin_prefetch(gB + 32, 0, 1);
      }
#endif
    }
#ifdef HAVE_ASYNC_LDS
    async_wait0();
#endif
    __syncthreads();

    Frag16 a[4], bf[2];
#pragma unroll
    for (int i = 0; i < 4; ++i) {
      int r = wm * 64 + i * 16 + (lane & 15);
      a[i].h[0] = *(const v8h*)&As[r][aoff];
      a[i].h[1] = *(const v8h*)&As[r][aoff + 16];
    }
#pragma unroll
    for (int j = 0; j < 2; ++j) {
      int r = wn * 32 + j * 16 + (lane & 15);
      bf[j].h[0] = *(const v8h*)&Bs[r][boff];
      bf[j].h[1] = *(const v8h*)&Bs[r][boff + 16];
    }
#pragma unroll
    for (int i = 0; i < 4; ++i)
#pragma unroll
      for (int j = 0; j < 2; ++j)
        acc[i][j] = __builtin_amdgcn_wmma_f32_16x16x32_f16(
            false, a[i].v, false, bf[j].v, (short)0, acc[i][j], false, false);
    __syncthreads();
  }

  // Epilogue.  C/D layout: VGPR p -> M = p + 8*(lane/16); N = lane%16.
#pragma unroll
  for (int i = 0; i < 4; ++i)
#pragma unroll
    for (int j = 0; j < 2; ++j)
#pragma unroll
      for (int p = 0; p < 8; ++p) {
        long row = m0 + wm * 64 + i * 16 + p + ((lane >> 4) << 3);
        long col = n0 + wn * 32 + j * 16 + (lane & 15);
        float v = acc[i][j][p] + bias[col];
        if constexpr (MODE == 0) {
          ((f16*)out)[row * N + col] = (f16)v;
        } else if constexpr (MODE == 1) {
          float g = 0.5f * v * (1.f + erff(v * 0.70710678118654752f));
          ((f16*)out)[row * N + col] = (f16)g;
        } else if constexpr (MODE == 2) {
          long bb = row >> 12;
          int rem = (int)(row & 4095);
          int wi = rem >> 6, t = rem & 63;
          int hs = (wi >> 3) * 8 + (t >> 3);
          int wsx = (wi & 7) * 8 + (t & 7);
          int h = (hs + 4) & 63, w = (wsx + 4) & 63;
          long orow = (bb << 12) + h * 64 + w;
          ((float*)out)[orow * N + col] = resid[orow * N + col] + v;
        } else {
          ((float*)out)[row * N + col] = resid[row * N + col] + v;
        }
      }
}

// ---------------------------------------------------------------------------
// Attention: one block (4 waves) per (window, head).  N=64 tokens, hd=32.
// Wave w handles query rows [16w, 16w+16).
// ---------------------------------------------------------------------------
#define VSTR 72   // LDS row stride (144B, multiple of 16B)

__global__ __launch_bounds__(128) void attn_kernel(const f16* __restrict__ QKV,
                                                   const float* __restrict__ biasHN,
                                                   const float* __restrict__ mask,
                                                   f16* __restrict__ O) {
  __shared__ f16 vT[32][VSTR];        // v transposed: [d][m]
  __shared__ f16 pS[4][16][VSTR];     // per-wave probabilities [row][m]

  int unit = blockIdx.x;
  int win = unit >> 4;
  int head = unit & 15;
  int wi = win & 63;
  int tid = threadIdx.x;
  int lane = tid & 31;
  int wave = tid >> 5;

  const f16* q0 = QKV + (long)win * 64 * QKVCH + head * 32;
  const f16* k0 = q0 + 512;
  const f16* v0 = q0 + 1024;

  // stage v^T into LDS
  for (int idx = tid; idx < 2048; idx += 128) {
    int m = idx >> 5, d = idx & 31;
    vT[d][m] = v0[(long)m * QKVCH + d];
  }
  __syncthreads();

  const int nb = wave * 16;
  const int aoff = (lane < 16) ? 0 : 8;
  const int boff = (lane < 16) ? 0 : 8;

  // Q fragment (16x32, K = head_dim)
  Frag16 qa;
  {
    long r = (long)(nb + (lane & 15)) * QKVCH;
    qa.h[0] = *(const v8h*)(q0 + r + aoff);
    qa.h[1] = *(const v8h*)(q0 + r + aoff + 16);
  }

  // scores: 4 WMMAs -> 16x64 strip
  v8f s[4];
#pragma unroll
  for (int mt = 0; mt < 4; ++mt) {
    Frag16 kb;
    long r = (long)(mt * 16 + (lane & 15)) * QKVCH;
    kb.h[0] = *(const v8h*)(k0 + r + boff);
    kb.h[1] = *(const v8h*)(k0 + r + boff + 16);
    v8f z = {0.f, 0.f, 0.f, 0.f, 0.f, 0.f, 0.f, 0.f};
    s[mt] = __builtin_amdgcn_wmma_f32_16x16x32_f16(
        false, qa.v, false, kb.v, (short)0, z, false, false);
  }

  // scale + rel-pos bias + shift mask + softmax (rows live in 16-lane halves)
  const float* bh = biasHN + head * 4096;
  const float* mk = mask + wi * 4096;
#pragma unroll
  for (int p = 0; p < 8; ++p) {
    int n = nb + p + ((lane >> 4) << 3);
    int mb = lane & 15;
    float t[4];
#pragma unroll
    for (int mt = 0; mt < 4; ++mt) {
      int mcol = mt * 16 + mb;
      t[mt] = s[mt][p] * ATT_SCALE + bh[n * 64 + mcol] + mk[n * 64 + mcol];
    }
    float mx = fmaxf(fmaxf(t[0], t[1]), fmaxf(t[2], t[3]));
#pragma unroll
    for (int m = 1; m < 16; m <<= 1) mx = fmaxf(mx, __shfl_xor(mx, m, 32));
    float sum = 0.f;
#pragma unroll
    for (int mt = 0; mt < 4; ++mt) {
      t[mt] = __expf(t[mt] - mx);
      sum += t[mt];
    }
#pragma unroll
    for (int m = 1; m < 16; m <<= 1) sum += __shfl_xor(sum, m, 32);
    float inv = 1.f / sum;
#pragma unroll
    for (int mt = 0; mt < 4; ++mt) s[mt][p] = t[mt] * inv;
  }

  // stash probabilities (f16) in per-wave LDS region
#pragma unroll
  for (int p = 0; p < 8; ++p) {
    int r = p + ((lane >> 4) << 3);
    int c = lane & 15;
#pragma unroll
    for (int mt = 0; mt < 4; ++mt) pS[wave][r][mt * 16 + c] = (f16)s[mt][p];
  }
  __syncthreads();

  // O = P (16x64) @ V (64x32): 2 k-steps x 2 n-tiles
  v8f oacc[2];
#pragma unroll
  for (int nt = 0; nt < 2; ++nt)
#pragma unroll
    for (int p = 0; p < 8; ++p) oacc[nt][p] = 0.f;

#pragma unroll
  for (int ks = 0; ks < 2; ++ks) {
    Frag16 pa;
    int r = lane & 15;
    int mo = ks * 32 + aoff;
    pa.h[0] = *(const v8h*)&pS[wave][r][mo];
    pa.h[1] = *(const v8h*)&pS[wave][r][mo + 16];
#pragma unroll
    for (int nt = 0; nt < 2; ++nt) {
      Frag16 vb;
      int d = nt * 16 + (lane & 15);
      int mo2 = ks * 32 + boff;
      vb.h[0] = *(const v8h*)&vT[d][mo2];
      vb.h[1] = *(const v8h*)&vT[d][mo2 + 16];
      oacc[nt] = __builtin_amdgcn_wmma_f32_16x16x32_f16(
          false, pa.v, false, vb.v, (short)0, oacc[nt], false, false);
    }
  }

#pragma unroll
  for (int nt = 0; nt < 2; ++nt)
#pragma unroll
    for (int p = 0; p < 8; ++p) {
      int n = nb + p + ((lane >> 4) << 3);
      int d = nt * 16 + (lane & 15);
      O[((long)win * 64 + n) * CCH + head * 32 + d] = (f16)oacc[nt][p];
    }
}

// ---------------------------------------------------------------------------
// Host launcher
// ---------------------------------------------------------------------------
extern "C" void kernel_launch(void* const* d_in, const int* in_sizes, int n_in,
                              void* d_out, int out_size, void* d_ws, size_t ws_size,
                              hipStream_t stream) {
  const float* x      = (const float*)d_in[0];
  const float* qkv_w  = (const float*)d_in[1];
  const float* qkv_b  = (const float*)d_in[2];
  const float* proj_w = (const float*)d_in[3];
  const float* proj_b = (const float*)d_in[4];
  const float* n1w    = (const float*)d_in[5];
  const float* n1b    = (const float*)d_in[6];
  const float* n2w    = (const float*)d_in[7];
  const float* n2b    = (const float*)d_in[8];
  const float* fc1_w  = (const float*)d_in[9];
  const float* fc1_b  = (const float*)d_in[10];
  const float* fc2_w  = (const float*)d_in[11];
  const float* fc2_b  = (const float*)d_in[12];
  const float* relt   = (const float*)d_in[13];
  const int*   reli   = (const int*)d_in[14];
  const float* amask  = (const float*)d_in[15];

  char* ws = (char*)d_ws;
  // Region layout (reused across stages):
  //   BIG : QKV f16 (192MB) then FC1 output f16 (256MB)
  //   H16 : LN output f16 (64MB), reused as attention output O
  f16*   BIG    = (f16*)ws;
  f16*   H16    = (f16*)(ws + 268435456ull);
  float* biasHN = (float*)(ws + 268435456ull + 67108864ull);
  f16*   qkv16  = (f16*)(ws + 268435456ull + 67108864ull + 262144ull);
  f16*   proj16 = qkv16 + 786432;
  f16*   fc116  = proj16 + 262144;
  f16*   fc216  = fc116 + 1048576;
  float* OUT1   = (float*)d_out;   // post-attention residual lives in d_out

  // weight conversion f32 -> f16
  cvt_f32_f16<<<3072, 256, 0, stream>>>(qkv_w, qkv16, 786432);
  cvt_f32_f16<<<1024, 256, 0, stream>>>(proj_w, proj16, 262144);
  cvt_f32_f16<<<4096, 256, 0, stream>>>(fc1_w, fc116, 1048576);
  cvt_f32_f16<<<4096, 256, 0, stream>>>(fc2_w, fc216, 1048576);

  // rel-pos bias gather
  bias_pre<<<256, 256, 0, stream>>>(relt, reli, biasHN);

  // LN1 + shift + window partition -> H16 (window space)
  ln_kernel<<<8192, 256, 0, stream>>>(x, n1w, n1b, H16, 1);

  // QKV GEMM: (65536x512) x (1536x512)^T -> BIG (f16)
  gemm_wmma<0><<<dim3(12, 512), 256, 0, stream>>>(H16, qkv16, qkv_b, nullptr,
                                                  BIG, ROWS, QKVCH, CCH);

  // attention per (window, head) -> H16 region (reused as O)
  attn_kernel<<<16384, 128, 0, stream>>>(BIG, biasHN, amask, H16);

  // proj GEMM + window reverse + roll + residual: OUT1 = x + proj(O)
  gemm_wmma<2><<<dim3(4, 512), 256, 0, stream>>>(H16, proj16, proj_b, x,
                                                 OUT1, ROWS, CCH, CCH);

  // LN2 -> H16
  ln_kernel<<<8192, 256, 0, stream>>>(OUT1, n2w, n2b, H16, 0);

  // FC1 GEMM + GELU -> BIG (f16)
  gemm_wmma<1><<<dim3(16, 512), 256, 0, stream>>>(H16, fc116, fc1_b, nullptr,
                                                  BIG, ROWS, HIDCH, CCH);

  // FC2 GEMM + residual -> d_out (in-place read of OUT1)
  gemm_wmma<3><<<dim3(4, 512), 256, 0, stream>>>(BIG, fc216, fc2_b, OUT1,
                                                 OUT1, ROWS, CCH, HIDCH);
}
